// AlignedSlotAttention_14843406975391
// MI455X (gfx1250) — matrix operand order repaired and checked by reading, not compiled, not run on hardware
//
#include <hip/hip_runtime.h>

// ---------------------------------------------------------------------------
// Types for CDNA5 WMMA (wave32)
// ---------------------------------------------------------------------------
typedef __attribute__((ext_vector_type(16))) __bf16 v16bf;
typedef __attribute__((ext_vector_type(8)))  float  v8f;
typedef unsigned short u16;
typedef __attribute__((ext_vector_type(8)))  unsigned short u16x8;

union Frag {
    v16bf v;
    u16   u[16];
    u16x8 h[2];
};

__device__ __forceinline__ u16 f2bf(float f) {
    unsigned int u = __float_as_uint(f);
    unsigned int r = u + 0x7FFFu + ((u >> 16) & 1u);   // round-to-nearest-even
    return (u16)(r >> 16);
}

__device__ __forceinline__ float gelu_tanh(float x) {
    const float k0 = 0.7978845608028654f, k1 = 0.044715f;
    return 0.5f * x * (1.0f + tanhf(k0 * (x + k1 * x * x * x)));
}

__device__ __forceinline__ v8f v8f_zero() {
    v8f z;
#pragma unroll
    for (int i = 0; i < 8; i++) z[i] = 0.0f;
    return z;
}

#define INV_SQRT_KD 0.17677669529663687f

// Problem constants
// B=128, N=256, D=128, H=4, KD=32, FF=512, L=4; M = B*N = 32768 rows.

// ---------------------------------------------------------------------------
// A-layout (activations): per 16x32 tile, 512 bf16 contiguous, element e of
// lane l = X[row = l%16][k = (e&7) + (e>>3)*16 + (l>>4)*8].  Tiles ordered
// row-tile-major, k-tile inner:  base = (mt*KT + kt)*512 + lane*16.
// B-layout (weights): per 32x16 tile, element e of lane l =
// W[k = (e&7) + (e>>3)*16 + (l>>4)*8][n = l%16].  base = (nt*KT + kt)*512 + lane*16.
// C/D fragment: vgpr g of lane l -> row = (l>>4)*8 + g, col = l%16.
// ---------------------------------------------------------------------------

// ---------------------------------------------------------------------------
// Weight conversion f32 row-major [K,N] -> bf16 B-layout
// ---------------------------------------------------------------------------
__global__ __launch_bounds__(256) void conv_w_kernel(const float* __restrict__ src,
                                                     u16* __restrict__ dst,
                                                     int K, int N) {
    int R   = blockIdx.x * 256 + threadIdx.x;          // run of 8 elements
    int KT  = K >> 5;
    int tile = R >> 6;
    int w    = R & 63;
    int l    = w >> 1;
    int hh   = w & 1;
    int nt   = tile / KT;
    int kt   = tile - nt * KT;
    int c    = nt * 16 + (l & 15);
    int kb   = kt * 32 + hh * 16 + (l >> 4) * 8;
    u16x8 o;
#pragma unroll
    for (int s = 0; s < 8; s++) o[s] = f2bf(src[(size_t)(kb + s) * N + c]);
    *(u16x8*)(dst + (size_t)tile * 512 + l * 16 + hh * 8) = o;
}

// ---------------------------------------------------------------------------
// Build x = concat(src[...,0:126], flag, pos) as f32 [M,128]
// ---------------------------------------------------------------------------
__global__ __launch_bounds__(128) void build_kernel(const float* __restrict__ src,
                                                    float* __restrict__ dst,
                                                    float flag) {
    int row = blockIdx.x;           // 0..32767
    int d   = threadIdx.x;          // 0..127
    int n   = row & 255;
    float v;
    if (d < 126)      v = src[(size_t)row * 126 + d];
    else if (d == 126) v = flag;
    else               v = -1.0f + 2.0f * (float)n / 255.0f;
    dst[(size_t)row * 128 + d] = v;
}

// ---------------------------------------------------------------------------
// LayerNorm over D=128 + pack to bf16 A-layout.  One block per 16-row tile.
// ---------------------------------------------------------------------------
__global__ __launch_bounds__(256) void ln_pack_kernel(const float* __restrict__ x,
                                                      const float* __restrict__ sc,
                                                      const float* __restrict__ of,
                                                      u16* __restrict__ out) {
    __shared__ float xs[16][128];
    __shared__ float red0[16][16];
    __shared__ float red1[16][16];
    __shared__ float mu[16], rs[16];

    int t  = threadIdx.x;
    int rt = blockIdx.x;                 // row-tile 0..2047
    int r  = t >> 4;
    int cb = (t & 15) * 8;
    const float* src = x + ((size_t)rt * 16 + r) * 128 + cb;
    float s = 0.f, s2 = 0.f;
#pragma unroll
    for (int i = 0; i < 8; i++) {
        float v = src[i];
        xs[r][cb + i] = v;
        s += v; s2 += v * v;
    }
    red0[r][t & 15] = s;
    red1[r][t & 15] = s2;
    __syncthreads();
    if (t < 16) {
        float a = 0.f, b = 0.f;
#pragma unroll
        for (int i = 0; i < 16; i++) { a += red0[t][i]; b += red1[t][i]; }
        float m   = a * (1.0f / 128.0f);
        float var = b * (1.0f / 128.0f) - m * m;
        mu[t] = m;
        rs[t] = rsqrtf(var + 1e-5f);
    }
    __syncthreads();
    // pack one run of 8 per thread: 4 k-tiles * 64 runs = 256 runs
    int kt = t >> 6, w = t & 63, la = w >> 1, hh = w & 1;
    int rr = la & 15, hi = la >> 4;
    int kb = kt * 32 + hh * 16 + hi * 8;
    float m = mu[rr], q = rs[rr];
    u16x8 o;
#pragma unroll
    for (int s_ = 0; s_ < 8; s_++) {
        int c = kb + s_;
        o[s_] = f2bf((xs[rr][c] - m) * q * sc[c] + of[c]);
    }
    *(u16x8*)(out + ((size_t)rt * 4 + kt) * 512 + la * 16 + hh * 8) = o;
}

// ---------------------------------------------------------------------------
// Generic bf16 WMMA GEMM: C[M, Ntot] = A[M, KT*32] @ W + bias (+epilogue)
// 256 threads = 8 waves; wave computes 16 rows x 128 cols.
// ---------------------------------------------------------------------------
enum { EPI_RESID = 0, EPI_ALAYOUT = 1, EPI_ROWMAJOR = 2 };

template <int KT, int EPI, bool GELU>
__global__ __launch_bounds__(256) void gemm_bf16_kernel(const u16* __restrict__ A,
                                                        const u16* __restrict__ W,
                                                        const float* __restrict__ bias,
                                                        float* __restrict__ resid,
                                                        u16* __restrict__ outb,
                                                        int Ntot) {
    int t    = threadIdx.x;
    int lane = t & 31;
    int wv   = t >> 5;
    int mt   = blockIdx.x * 8 + wv;
    int nt0  = blockIdx.y * 8;
    int col  = lane & 15;
    int hi   = lane >> 4;

    v8f acc[8];
#pragma unroll
    for (int j = 0; j < 8; j++) acc[j] = v8f_zero();

    const u16* Ap = A + (size_t)mt * KT * 512 + lane * 16;
#pragma unroll
    for (int kt = 0; kt < KT; kt++) {
        v16bf a = *(const v16bf*)(Ap + kt * 512);
#pragma unroll
        for (int j = 0; j < 8; j++) {
            v16bf b = *(const v16bf*)(W + ((size_t)(nt0 + j) * KT + kt) * 512 + lane * 16);
            acc[j] = __builtin_amdgcn_wmma_f32_16x16x32_bf16(
                false, a, false, b, (short)0, acc[j], false, false);
        }
    }

    if constexpr (EPI == EPI_RESID) {
#pragma unroll
        for (int j = 0; j < 8; j++) {
            int   c  = (nt0 + j) * 16 + col;
            float bb = bias[c];
#pragma unroll
            for (int g = 0; g < 8; g++) {
                size_t idx = (size_t)(mt * 16 + hi * 8 + g) * Ntot + c;
                resid[idx] += acc[j][g] + bb;
            }
        }
    } else {
        __shared__ u16 stage[128 * 128];   // 32 KB
#pragma unroll
        for (int j = 0; j < 8; j++) {
            int   c  = j * 16 + col;
            float bb = bias[nt0 * 16 + c];
#pragma unroll
            for (int g = 0; g < 8; g++) {
                float v = acc[j][g] + bb;
                if (GELU) v = gelu_tanh(v);
                stage[(wv * 16 + hi * 8 + g) * 128 + c] = f2bf(v);
            }
        }
        __syncthreads();
        if constexpr (EPI == EPI_ALAYOUT) {
            int KTtot = Ntot >> 5;
#pragma unroll
            for (int it = 0; it < 8; it++) {
                int R    = it * 256 + t;
                int tile = R >> 6, w = R & 63, la = w >> 1, hh = w & 1;
                int lmt  = tile >> 2, lkt = tile & 3;
                int rl   = la & 15, h2 = la >> 4;
                int kl   = lkt * 32 + hh * 16 + h2 * 8;
                u16x8 o  = *(const u16x8*)&stage[(lmt * 16 + rl) * 128 + kl];
                size_t base = ((size_t)(blockIdx.x * 8 + lmt) * KTtot +
                               (blockIdx.y * 4 + lkt)) * 512;
                *(u16x8*)(outb + base + la * 16 + hh * 8) = o;
            }
        } else { // EPI_ROWMAJOR
            int lr = t >> 1, cb = (t & 1) * 64;
            size_t grow = (size_t)blockIdx.x * 128 + lr;
            int    gc   = blockIdx.y * 128 + cb;
#pragma unroll
            for (int i = 0; i < 8; i++)
                *(u16x8*)(outb + grow * Ntot + gc + 8 * i) =
                    *(const u16x8*)&stage[lr * 128 + cb + 8 * i];
        }
    }
}

// ---------------------------------------------------------------------------
// Attention (layers 0..2): softmax(Q Kᵀ)*inv_sqrt_kd @ V for one (b,h,16-row
// strip) per wave; 2 waves/block sharing the V slice in LDS.
// q,k: bf16 A-layout (k-tile == head).  v: bf16 row-major [M,128].
// attn out: bf16 A-layout.
// ---------------------------------------------------------------------------
__global__ __launch_bounds__(64) void attn_mid_kernel(const u16* __restrict__ q,
                                                      const u16* __restrict__ k,
                                                      const u16* __restrict__ v,
                                                      u16* __restrict__ attn) {
    __shared__ u16 Vs[256 * 32];       // 16 KB (head slice of V)
    __shared__ u16 Ws[2][16 * 256];    // 16 KB (per-wave w staging)

    int t    = threadIdx.x;
    int lane = t & 31;
    int wv   = t >> 5;
    int bx   = blockIdx.x;             // B*H*8 = 4096
    int bh   = bx >> 3;
    int b    = bh >> 2, h = bh & 3;
    int mt   = (bx & 7) * 2 + wv;
    int r    = lane & 15, hi = lane >> 4;

    // cooperative V slice load (coalesced 16B chunks)
#pragma unroll
    for (int i = 0; i < 16; i++) {
        int c  = t + 64 * i;
        int eb = c * 8;
        int T  = eb >> 5, d = eb & 31;
        *(u16x8*)&Vs[eb] = *(const u16x8*)&v[((size_t)b * 256 + T) * 128 + h * 32 + d];
    }
    __syncthreads();

    // Q fragment
    Frag a;
    a.v = *(const v16bf*)(q + ((size_t)(b * 16 + mt) * 4 + h) * 512 + lane * 16);

    // logits strip: 16 tiles of 16x16
    v8f acc[16];
#pragma unroll
    for (int Tt = 0; Tt < 16; Tt++) acc[Tt] = v8f_zero();
#pragma unroll
    for (int Tt = 0; Tt < 16; Tt++) {
        const u16* kb = k + ((size_t)(b * 16 + Tt) * 4 + h) * 512;
        Frag bb;
        bb.h[0] = *(const u16x8*)&kb[lane * 16];
        bb.h[1] = *(const u16x8*)&kb[lane * 16 + 8];
        acc[Tt] = __builtin_amdgcn_wmma_f32_16x16x32_bf16(
            false, a.v, false, bb.v, (short)0, acc[Tt], false, false);
    }

    // softmax per row (16 lanes of a half hold one row across 16 tiles)
#pragma unroll
    for (int g = 0; g < 8; g++) {
        float m = -1e30f;
#pragma unroll
        for (int Tt = 0; Tt < 16; Tt++) m = fmaxf(m, acc[Tt][g]);
#pragma unroll
        for (int d = 8; d >= 1; d >>= 1) m = fmaxf(m, __shfl_xor(m, d, 32));
        float s = 0.f;
#pragma unroll
        for (int Tt = 0; Tt < 16; Tt++) {
            float e = __expf(acc[Tt][g] - m);
            acc[Tt][g] = e;
            s += e;
        }
#pragma unroll
        for (int d = 8; d >= 1; d >>= 1) s += __shfl_xor(s, d, 32);
        float scf = INV_SQRT_KD / s;
        int row = hi * 8 + g;
#pragma unroll
        for (int Tt = 0; Tt < 16; Tt++)
            Ws[wv][row * 256 + Tt * 16 + r] = f2bf(acc[Tt][g] * scf);
    }
    __syncthreads();

    // attn = w @ V   (8 k-steps of 32, 2 column tiles)
    v8f oacc[2];
    oacc[0] = v8f_zero(); oacc[1] = v8f_zero();
#pragma unroll
    for (int kt = 0; kt < 8; kt++) {
        Frag af;
        af.h[0] = *(const u16x8*)&Ws[wv][r * 256 + kt * 32 + hi * 8];
        af.h[1] = *(const u16x8*)&Ws[wv][r * 256 + kt * 32 + hi * 8 + 16];
#pragma unroll
        for (int nt = 0; nt < 2; nt++) {
            Frag bb;
#pragma unroll
            for (int e = 0; e < 16; e++)
                bb.u[e] = Vs[(kt * 32 + hi * 8 + e + (e & 8)) * 32 + nt * 16 + r];
            oacc[nt] = __builtin_amdgcn_wmma_f32_16x16x32_bf16(
                false, af.v, false, bb.v, (short)0, oacc[nt], false, false);
        }
    }
    __syncthreads();

    // restage 16x32 result and pack A-layout (k-tile index == head)
#pragma unroll
    for (int nt = 0; nt < 2; nt++)
#pragma unroll
        for (int g = 0; g < 8; g++)
            Ws[wv][(hi * 8 + g) * 32 + nt * 16 + r] = f2bf(oacc[nt][g]);
    __syncthreads();

    u16* dst = attn + ((size_t)(b * 16 + mt) * 4 + h) * 512;
#pragma unroll
    for (int hh = 0; hh < 2; hh++) {
        u16x8 o;
#pragma unroll
        for (int s = 0; s < 8; s++)
            o[s] = Ws[wv][r * 32 + hh * 16 + hi * 8 + s];
        *(u16x8*)&dst[lane * 16 + hh * 8] = o;
    }
}

// ---------------------------------------------------------------------------
// Final attention: P = sum_h softmax(QKᵀ)*inv_sqrt_kd ; writes S = exp(P).
// One wave per (b, 16-row strip); 2 waves/block.
// ---------------------------------------------------------------------------
__global__ __launch_bounds__(64) void attn_final_kernel(const u16* __restrict__ q,
                                                        const u16* __restrict__ k,
                                                        float* __restrict__ S) {
    __shared__ float Ps[2][16 * 256];  // 32 KB

    int t    = threadIdx.x;
    int lane = t & 31;
    int wv   = t >> 5;
    int bx   = blockIdx.x;             // B*8 = 1024
    int b    = bx >> 3;
    int mt   = (bx & 7) * 2 + wv;
    int r    = lane & 15, hi = lane >> 4;

    float4 z4 = make_float4(0.f, 0.f, 0.f, 0.f);
#pragma unroll
    for (int i = 0; i < 32; i++)
        *(float4*)&Ps[wv][(i * 32 + lane) * 4] = z4;

    for (int h = 0; h < 4; h++) {
        Frag a;
        a.v = *(const v16bf*)(q + ((size_t)(b * 16 + mt) * 4 + h) * 512 + lane * 16);
        v8f acc[16];
#pragma unroll
        for (int Tt = 0; Tt < 16; Tt++) acc[Tt] = v8f_zero();
#pragma unroll
        for (int Tt = 0; Tt < 16; Tt++) {
            const u16* kb = k + ((size_t)(b * 16 + Tt) * 4 + h) * 512;
            Frag bb;
            bb.h[0] = *(const u16x8*)&kb[lane * 16];
            bb.h[1] = *(const u16x8*)&kb[lane * 16 + 8];
            acc[Tt] = __builtin_amdgcn_wmma_f32_16x16x32_bf16(
                false, a.v, false, bb.v, (short)0, acc[Tt], false, false);
        }
#pragma unroll
        for (int g = 0; g < 8; g++) {
            float m = -1e30f;
#pragma unroll
            for (int Tt = 0; Tt < 16; Tt++) m = fmaxf(m, acc[Tt][g]);
#pragma unroll
            for (int d = 8; d >= 1; d >>= 1) m = fmaxf(m, __shfl_xor(m, d, 32));
            float s = 0.f;
#pragma unroll
            for (int Tt = 0; Tt < 16; Tt++) {
                float e = __expf(acc[Tt][g] - m);
                acc[Tt][g] = e;
                s += e;
            }
#pragma unroll
            for (int d = 8; d >= 1; d >>= 1) s += __shfl_xor(s, d, 32);
            float scf = INV_SQRT_KD / s;
            int row = hi * 8 + g;
#pragma unroll
            for (int Tt = 0; Tt < 16; Tt++)
                Ps[wv][row * 256 + Tt * 16 + r] += acc[Tt][g] * scf;
        }
    }

    float* dst = S + (size_t)b * 65536 + (size_t)mt * 16 * 256;
#pragma unroll
    for (int i = 0; i < 32; i++) {
        int idx = (i * 32 + lane) * 4;
        float4 p = *(float4*)&Ps[wv][idx];
        float4 o = make_float4(__expf(p.x), __expf(p.y), __expf(p.z), __expf(p.w));
        *(float4*)&dst[idx] = o;
    }
}

// ---------------------------------------------------------------------------
// Sinkhorn normalizations (in-place on S [B,256,256])
// ---------------------------------------------------------------------------
__global__ __launch_bounds__(256) void row_norm_kernel(float* __restrict__ S) {
    int t    = threadIdx.x;
    int lane = t & 31;
    int wv   = t >> 5;
    size_t row = (size_t)blockIdx.x * 8 + wv;       // 0..32767
    float* p = S + row * 256 + lane * 8;
    float4 a = *(float4*)p;
    float4 c = *(float4*)(p + 4);
    float s = a.x + a.y + a.z + a.w + c.x + c.y + c.z + c.w;
#pragma unroll
    for (int d = 16; d >= 1; d >>= 1) s += __shfl_xor(s, d, 32);
    float inv = 1.0f / s;
    a.x *= inv; a.y *= inv; a.z *= inv; a.w *= inv;
    c.x *= inv; c.y *= inv; c.z *= inv; c.w *= inv;
    *(float4*)p = a;
    *(float4*)(p + 4) = c;
}

__global__ __launch_bounds__(256) void col_norm_kernel(float* __restrict__ S) {
    int b = blockIdx.x;
    int j = threadIdx.x;
    float* p = S + (size_t)b * 65536 + j;
    float s = 0.f;
    for (int i = 0; i < 256; i++) s += p[i * 256];
    float inv = 1.0f / s;
    for (int i = 0; i < 256; i++) p[i * 256] *= inv;
}

// ---------------------------------------------------------------------------
// Host orchestration
// ---------------------------------------------------------------------------
extern "C" void kernel_launch(void* const* d_in, const int* in_sizes, int n_in,
                              void* d_out, int out_size, void* d_ws, size_t ws_size,
                              hipStream_t stream) {
    (void)in_sizes; (void)n_in; (void)out_size; (void)ws_size;

    const float* in_xd  = (const float*)d_in[0];
    const float* in_xo  = (const float*)d_in[1];
    const float* ln1d_s = (const float*)d_in[2];
    const float* ln1d_o = (const float*)d_in[3];
    const float* ln1o_s = (const float*)d_in[4];
    const float* ln1o_o = (const float*)d_in[5];
    const float* ln2_s  = (const float*)d_in[6];
    const float* ln2_o  = (const float*)d_in[7];
    const float* wq  = (const float*)d_in[8];   const float* bq  = (const float*)d_in[9];
    const float* wk  = (const float*)d_in[10];  const float* bk  = (const float*)d_in[11];
    const float* wvw = (const float*)d_in[12];  const float* bv  = (const float*)d_in[13];
    const float* wo  = (const float*)d_in[14];  const float* bo  = (const float*)d_in[15];
    const float* w1o = (const float*)d_in[16];  const float* b1o = (const float*)d_in[17];
    const float* w2o = (const float*)d_in[18];  const float* b2o = (const float*)d_in[19];
    const float* w1d = (const float*)d_in[20];  const float* b1d = (const float*)d_in[21];
    const float* w2d = (const float*)d_in[22];  const float* b2d = (const float*)d_in[23];
    float* Sout = (float*)d_out;

    // workspace layout (bytes); total ~129 MB
    char* ws = (char*)d_ws;
    constexpr size_t OFF_XDYN = 0;
    constexpr size_t OFF_XOBS = OFF_XDYN + 16777216;
    constexpr size_t OFF_HD   = OFF_XOBS + 16777216;
    constexpr size_t OFF_HO   = OFF_HD   + 8388608;
    constexpr size_t OFF_HO2  = OFF_HO   + 8388608;
    constexpr size_t OFF_Q    = OFF_HO2  + 8388608;
    constexpr size_t OFF_K    = OFF_Q    + 8388608;
    constexpr size_t OFF_V    = OFF_K    + 8388608;
    constexpr size_t OFF_ATTN = OFF_V    + 8388608;
    constexpr size_t OFF_T    = OFF_ATTN + 8388608;
    constexpr size_t OFF_WC   = OFF_T    + 33554432;

    float* xdyn = (float*)(ws + OFF_XDYN);
    float* xobs = (float*)(ws + OFF_XOBS);
    u16* hd   = (u16*)(ws + OFF_HD);
    u16* ho   = (u16*)(ws + OFF_HO);
    u16* ho2  = (u16*)(ws + OFF_HO2);
    u16* qb   = (u16*)(ws + OFF_Q);
    u16* kb   = (u16*)(ws + OFF_K);
    u16* vb   = (u16*)(ws + OFF_V);
    u16* ab   = (u16*)(ws + OFF_ATTN);
    u16* tb   = (u16*)(ws + OFF_T);
    u16* wc   = (u16*)(ws + OFF_WC);

    // convert all weights to bf16 B-layout (held in L2 across the pass)
    for (int l = 0; l < 4; l++) {
        u16* base = wc + (size_t)l * 327680;
        conv_w_kernel<<<8,  256, 0, stream>>>(wq  + (size_t)l * 16384, base +      0, 128, 128);
        conv_w_kernel<<<8,  256, 0, stream>>>(wk  + (size_t)l * 16384, base +  16384, 128, 128);
        conv_w_kernel<<<8,  256, 0, stream>>>(wvw + (size_t)l * 16384, base +  32768, 128, 128);
        conv_w_kernel<<<8,  256, 0, stream>>>(wo  + (size_t)l * 16384, base +  49152, 128, 128);
        conv_w_kernel<<<32, 256, 0, stream>>>(w1o + (size_t)l * 65536, base +  65536, 128, 512);
        conv_w_kernel<<<32, 256, 0, stream>>>(w2o + (size_t)l * 65536, base + 131072, 512, 128);
        conv_w_kernel<<<32, 256, 0, stream>>>(w1d + (size_t)l * 65536, base + 196608, 128, 512);
        conv_w_kernel<<<32, 256, 0, stream>>>(w2d + (size_t)l * 65536, base + 262144, 512, 128);
    }

    build_kernel<<<32768, 128, 0, stream>>>(in_xd, xdyn, -1.0f);
    build_kernel<<<32768, 128, 0, stream>>>(in_xo, xobs,  1.0f);

    for (int l = 0; l < 4; l++) {
        u16* Wb = wc + (size_t)l * 327680;
        ln_pack_kernel<<<2048, 256, 0, stream>>>(xdyn, ln1d_s + l * 128, ln1d_o + l * 128, hd);
        ln_pack_kernel<<<2048, 256, 0, stream>>>(xobs, ln1o_s + l * 128, ln1o_o + l * 128, ho);
        gemm_bf16_kernel<4, EPI_ALAYOUT, false><<<dim3(256, 1), 256, 0, stream>>>(
            hd, Wb + 0, bq + l * 128, nullptr, qb, 128);
        gemm_bf16_kernel<4, EPI_ALAYOUT, false><<<dim3(256, 1), 256, 0, stream>>>(
            ho, Wb + 16384, bk + l * 128, nullptr, kb, 128);
        if (l < 3) {
            gemm_bf16_kernel<4, EPI_ROWMAJOR, false><<<dim3(256, 1), 256, 0, stream>>>(
                ho, Wb + 32768, bv + l * 128, nullptr, vb, 128);
            attn_mid_kernel<<<4096, 64, 0, stream>>>(qb, kb, vb, ab);
            gemm_bf16_kernel<4, EPI_RESID, false><<<dim3(256, 1), 256, 0, stream>>>(
                ab, Wb + 49152, bo + l * 128, xobs, nullptr, 128);
            ln_pack_kernel<<<2048, 256, 0, stream>>>(xobs, ln2_s + l * 128, ln2_o + l * 128, ho2);
            gemm_bf16_kernel<4, EPI_ALAYOUT, true><<<dim3(256, 4), 256, 0, stream>>>(
                ho2, Wb + 65536, b1o + l * 512, nullptr, tb, 512);
            gemm_bf16_kernel<16, EPI_RESID, false><<<dim3(256, 1), 256, 0, stream>>>(
                tb, Wb + 131072, b2o + l * 128, xobs, nullptr, 128);
            gemm_bf16_kernel<4, EPI_ALAYOUT, true><<<dim3(256, 4), 256, 0, stream>>>(
                hd, Wb + 196608, b1d + l * 512, nullptr, tb, 512);
            gemm_bf16_kernel<16, EPI_RESID, false><<<dim3(256, 1), 256, 0, stream>>>(
                tb, Wb + 262144, b2d + l * 128, xdyn, nullptr, 128);
        } else {
            attn_final_kernel<<<1024, 64, 0, stream>>>(qb, kb, Sout);
        }
    }

    for (int it = 0; it < 5; it++) {
        row_norm_kernel<<<4096, 256, 0, stream>>>(Sout);
        col_norm_kernel<<<128,  256, 0, stream>>>(Sout);
    }
}
